// NequIPConvolution_11390253269438
// MI455X (gfx1250) — matrix operand order repaired and checked by reading, not compiled, not run on hardware
//
#include <hip/hip_runtime.h>

typedef __attribute__((ext_vector_type(16))) _Float16 v16h;
typedef __attribute__((ext_vector_type(8)))  _Float16 v8h;
typedef __attribute__((ext_vector_type(8)))  float    v8f;
typedef unsigned int u32x4 __attribute__((ext_vector_type(4)));
typedef int          i32x8 __attribute__((ext_vector_type(8)));
typedef int          i32x4 __attribute__((ext_vector_type(4)));

#define N_NODES 50000
#define N_EDGES 800000

// ---------------- packed f16 weight-fragment offsets (units: halfs) -------
// fragment = 512 halfs: [lane(32)][elem(16)], B layout: k = kt*32 + (i&7) + 16*(i>>3) + 8*half, n = nt*16 + (lane&15)
#define OFF_L1W0 0        // (32x32)   1x2 frags
#define OFF_L1W1 1024     // (32x32)   1x2
#define OFF_SCWS 2048     // (512x32) 16x2
#define OFF_SCWG 18432    // (512x32) 16x2
#define OFF_SCWV 34816    // (512x32) 16x2
#define OFF_FCW0 51200    // (8->32 pad x64) 1x4
#define OFF_FCW1 53248    // (64x64)  2x4
#define OFF_FCW2 57344    // (64x128) 2x8
#define OFF_L2W0 65536    // (64x64)  2x4
#define OFF_L2W1 69632    // (64x32)  2x2
#define PK_HALFS 71680

#define NODE_WLDS_HALFS 51200   // [OFF_L1W0, OFF_FCW0)  : 100KB
#define EDGE_WLDS_HALFS 14336   // [OFF_FCW0, OFF_L2W0)  : 28KB
#define FIN_WLDS_HALFS  6144    // [OFF_L2W0, PK_HALFS)  : 12KB

// ---------------- float workspace (after packed weights), units: floats ---
#define H0_F   0
#define H1_F   (H0_F  + N_NODES*32)
#define SCS_F  (H1_F  + N_NODES*96)
#define SCG_F  (SCS_F + N_NODES*32)
#define SCV_F  (SCG_F + N_NODES*32)
#define S0_F   (SCV_F + N_NODES*96)
#define S1_F   (S0_F  + N_NODES*64)

__device__ __forceinline__ v8f vzero8f() {
  v8f z;
#pragma unroll
  for (int i = 0; i < 8; ++i) z[i] = 0.f;
  return z;
}

__device__ __forceinline__ v8f wmma32(v16h a, v16h b, v8f c) {
  return __builtin_amdgcn_wmma_f32_16x16x32_f16(false, a, false, b, (short)0, c,
                                                false, false);
}

// B-fragment load (works for both global packed buffer and LDS-staged copy)
__device__ __forceinline__ v16h ldB(const _Float16* pk, int off, int frag,
                                    int lane) {
  return *(const v16h*)(pk + off + frag * 512 + lane * 16);
}

__device__ __forceinline__ float silu(float x) {
  return x * (1.0f / (1.0f + __expf(-x)));
}

// Wave-private LDS fence: all DS writes from this wave complete before reads.
__device__ __forceinline__ void wave_lds_fence() {
  __builtin_amdgcn_wave_barrier();
  asm volatile("s_wait_dscnt 0" ::: "memory");
  __builtin_amdgcn_wave_barrier();
}

// A-fragment k index for element i: (i&7) + (i>=8 ? 16 : 0) + 8*half
__device__ __forceinline__ int kkmap(int i, int h8) {
  return (i & 7) + ((i >> 3) << 4) + h8;
}

__device__ __forceinline__ void atomAddF(float* p, float v) {
  unsafeAtomicAdd(p, v);   // hardware global_atomic_add_f32 (no CAS loop)
}

// ---- Tensor Data Mover: 1-D bulk copy global->LDS (bytes % 8 == 0) -------
// D# per CDNA5 ISA ch.8: group0 = {count|lds_addr|global_addr|type=2},
// group1 = {data_size=8B, tensor_dim0/tile_dim0 = bytes/8, stride = same}.
__device__ __forceinline__ void tdm_load_to_lds(void* lds, const void* gsrc,
                                                unsigned bytes) {
  unsigned lds_off = (unsigned)(unsigned long long)lds; // low 32b = LDS offset
  unsigned long long ga = (unsigned long long)gsrc;
  unsigned units = bytes >> 3;                          // 8-byte elements
  u32x4 g0;
  g0[0] = 1u;                                            // count = 1
  g0[1] = lds_off;                                       // lds_addr
  g0[2] = (unsigned)ga;                                  // global_addr[31:0]
  g0[3] = (unsigned)((ga >> 32) & 0x1FFFFFFu) | (2u << 30); // addr hi | type=2
  i32x8 g1;
  g1[0] = (int)(3u << 16);                     // workgroup_mask=0, data_size=3 (8B)
  g1[1] = (int)((units & 0xFFFFu) << 16);      // tensor_dim0 lo16 @ [63:48]
  g1[2] = (int)((units >> 16) & 0xFFFFu) | (1 << 16); // dim0 hi16, tensor_dim1=1
  g1[3] = (int)((units & 0xFFFFu) << 16);      // tile_dim0 @ [127:112]
  g1[4] = 0;                                   // tile_dim1=0, tile_dim2=0
  g1[5] = (int)units;                          // tensor_dim0_stride lo32
  g1[6] = 0;
  g1[7] = 0;
  i32x4 z4;
  z4[0] = z4[1] = z4[2] = z4[3] = 0;
#if __has_include(<hip/amd_detail/amd_gfx1250_TDM.h>)
  i32x8 z8;
#pragma unroll
  for (int i = 0; i < 8; ++i) z8[i] = 0;
  __builtin_amdgcn_tensor_load_to_lds(g0, g1, z4, z4, z8, 0);
#else
  __builtin_amdgcn_tensor_load_to_lds(g0, g1, z4, z4, 0);
#endif
}

// stage [pk+off_halfs, +n_halfs) into LDS once per block (wave 0 issues DMA)
__device__ __forceinline__ void stage_weights(_Float16* wlds,
                                              const _Float16* pk,
                                              int off_halfs, int n_halfs,
                                              int wid) {
  if (wid == 0) {
    tdm_load_to_lds((void*)wlds, (const void*)(pk + off_halfs),
                    (unsigned)n_halfs * 2u);
    __builtin_amdgcn_s_wait_tensorcnt(0);
  }
  __syncthreads();
}

// ======================== weight pack kernel ==============================
__device__ void pack_one(const float* __restrict__ W, int K, int N, int Kpad,
                         float scale, _Float16* __restrict__ dst) {
  int ktiles = Kpad >> 5, ntiles = N >> 4;
  int total = ktiles * ntiles * 512;
  for (int idx = threadIdx.x + blockIdx.x * blockDim.x; idx < total;
       idx += blockDim.x * gridDim.x) {
    int frag = idx >> 9;
    int lane = (idx >> 4) & 31;
    int i    = idx & 15;
    int kt = frag / ntiles, nt = frag - kt * ntiles;
    int half = lane >> 4, col = lane & 15;
    int k = kt * 32 + (i & 7) + ((i >> 3) << 4) + 8 * half;
    int n = nt * 16 + col;
    float v = (k < K) ? W[k * N + n] * scale : 0.f;
    dst[idx] = (_Float16)v;
  }
}

__global__ __launch_bounds__(256) void pack_weights_kernel(
    const float* __restrict__ l1w0, const float* __restrict__ l1w1,
    const float* __restrict__ fcw0, const float* __restrict__ fcw1,
    const float* __restrict__ fcw2, const float* __restrict__ scws,
    const float* __restrict__ scwg, const float* __restrict__ scwv,
    const float* __restrict__ l2w0, const float* __restrict__ l2w1,
    _Float16* __restrict__ pk) {
  const float RS_M  = 0.17677669529663689f;   // 1/sqrt(32)
  const float RS_NB = 0.35355339059327373f;   // 1/sqrt(8)
  const float RS_RH = 0.125f;                 // 1/sqrt(64)
  const float RS_SC = 0.04419417382415922f;   // 1/sqrt(512)
  const float RS_L2 = 0.125f * (1.0f / 16.0f);// l2 norm * inv_nb folded
  pack_one(l1w0, 32,  32, 32,  RS_M,  pk + OFF_L1W0);
  pack_one(l1w1, 32,  32, 32,  RS_M,  pk + OFF_L1W1);
  pack_one(scws, 512, 32, 512, RS_SC, pk + OFF_SCWS);
  pack_one(scwg, 512, 32, 512, RS_SC, pk + OFF_SCWG);
  pack_one(scwv, 512, 32, 512, RS_SC, pk + OFF_SCWV);
  pack_one(fcw0, 8,   64, 32,  RS_NB, pk + OFF_FCW0);
  pack_one(fcw1, 64,  64, 64,  RS_RH, pk + OFF_FCW1);
  pack_one(fcw2, 64, 128, 64,  RS_RH, pk + OFF_FCW2);
  pack_one(l2w0, 64,  64, 64,  RS_L2, pk + OFF_L2W0);
  pack_one(l2w1, 64,  32, 64,  RS_L2, pk + OFF_L2W1);
}

// ======================== node prep: lin1 + self-connection ===============
// One wave per 16 nodes. sc einsums done as (x⊗z)[16x512] @ W[512x32] GEMMs.
// Weights TDM-staged to LDS once per block; B fragments come from ds loads.
__global__ __launch_bounds__(128) void node_prep_kernel(
    const float* __restrict__ nf, const float* __restrict__ zattr,
    const _Float16* __restrict__ pk, float* __restrict__ ws) {
  __shared__ __align__(32) _Float16 wlds[NODE_WLDS_HALFS];  // 100KB
  int lane = threadIdx.x & 31;
  int wid  = threadIdx.x >> 5;
  int tile = blockIdx.x * 4 + wid;

  stage_weights(wlds, pk, OFF_L1W0, NODE_WLDS_HALFS, wid);
  if (tile * 16 >= N_NODES) return;

  int r = lane & 15;
  int half = lane >> 4;
  int h8 = half * 8;
  int node = tile * 16 + r;
  const float* nfr = nf + (size_t)node * 128;

  // z row of my node, kept in VGPRs
  float zr[16];
  const float4* zp = (const float4*)(zattr + (size_t)node * 16);
#pragma unroll
  for (int j = 0; j < 4; ++j) {
    float4 t = zp[j];
    zr[4 * j + 0] = t.x; zr[4 * j + 1] = t.y;
    zr[4 * j + 2] = t.z; zr[4 * j + 3] = t.w;
  }

  float* h0  = ws + H0_F;
  float* h1  = ws + H1_F;
  float* scs = ws + SCS_F;
  float* scg = ws + SCG_F;
  float* scv = ws + SCV_F;

  // ---- pass over x0: sc_s, sc_g (K=512 kron GEMM) ----
  v8f as[2], ag[2];
  as[0] = vzero8f(); as[1] = vzero8f();
  ag[0] = vzero8f(); ag[1] = vzero8f();
  for (int t = 0; t < 16; ++t) {
    float su0 = nfr[2 * t], su1 = nfr[2 * t + 1];
    v16h a;
#pragma unroll
    for (int i = 0; i < 16; ++i) {
      float sv = (i < 8) ? su0 : su1;           // u = 2t (+1 for hi elems)
      a[i] = (_Float16)(sv * zr[(i & 7) + h8]); // v = (i&7)+8*half
    }
#pragma unroll
    for (int nt = 0; nt < 2; ++nt) {
      as[nt] = wmma32(a, ldB(wlds, OFF_SCWS, t * 2 + nt, lane), as[nt]);
      ag[nt] = wmma32(a, ldB(wlds, OFF_SCWG, t * 2 + nt, lane), ag[nt]);
    }
  }
#pragma unroll
  for (int nt = 0; nt < 2; ++nt)
#pragma unroll
    for (int i = 0; i < 8; ++i) {
      int row = tile * 16 + i + h8;
      scs[(size_t)row * 32 + nt * 16 + r] = as[nt][i];
      scg[(size_t)row * 32 + nt * 16 + r] = ag[nt][i];
    }

  // ---- lin1: h0 = x0 @ w0 (K=32) ----
  {
    v16h ax;
#pragma unroll
    for (int i = 0; i < 16; ++i) ax[i] = (_Float16)nfr[kkmap(i, h8)];
    v8f a0 = wmma32(ax, ldB(wlds, OFF_L1W0, 0, lane), vzero8f());
    v8f a1 = wmma32(ax, ldB(wlds, OFF_L1W0, 1, lane), vzero8f());
#pragma unroll
    for (int i = 0; i < 8; ++i) {
      int row = tile * 16 + i + h8;
      h0[(size_t)row * 32 + r]      = a0[i];
      h0[(size_t)row * 32 + 16 + r] = a1[i];
    }
  }

  // ---- per vector component m: lin1 h1 and sc_v ----
  for (int m = 0; m < 3; ++m) {
    v16h axm;
#pragma unroll
    for (int i = 0; i < 16; ++i)
      axm[i] = (_Float16)nfr[32 + kkmap(i, h8) * 3 + m];
    v8f b0 = wmma32(axm, ldB(wlds, OFF_L1W1, 0, lane), vzero8f());
    v8f b1 = wmma32(axm, ldB(wlds, OFF_L1W1, 1, lane), vzero8f());
#pragma unroll
    for (int i = 0; i < 8; ++i) {
      int row = tile * 16 + i + h8;
      h1[(size_t)row * 96 + r * 3 + m]        = b0[i];
      h1[(size_t)row * 96 + (16 + r) * 3 + m] = b1[i];
    }

    v8f av[2];
    av[0] = vzero8f(); av[1] = vzero8f();
    for (int t = 0; t < 16; ++t) {
      float su0 = nfr[32 + 6 * t + m];
      float su1 = nfr[32 + 6 * t + 3 + m];
      v16h a;
#pragma unroll
      for (int i = 0; i < 16; ++i) {
        float sv = (i < 8) ? su0 : su1;
        a[i] = (_Float16)(sv * zr[(i & 7) + h8]);
      }
#pragma unroll
      for (int nt = 0; nt < 2; ++nt)
        av[nt] = wmma32(a, ldB(wlds, OFF_SCWV, t * 2 + nt, lane), av[nt]);
    }
#pragma unroll
    for (int nt = 0; nt < 2; ++nt)
#pragma unroll
      for (int i = 0; i < 8; ++i) {
        int row = tile * 16 + i + h8;
        scv[(size_t)row * 96 + (nt * 16 + r) * 3 + m] = av[nt][i];
      }
  }
}

// ======================== edge kernel: MLP + tensor product + scatter =====
// One wave per 16 edges; WMMA MLP 8->64->64->128, then atomics into s0/s1.
__global__ __launch_bounds__(128) void edge_kernel(
    const float* __restrict__ esh, const float* __restrict__ emb,
    const int* __restrict__ esrc, const int* __restrict__ edst,
    const _Float16* __restrict__ pk, float* __restrict__ ws) {
  __shared__ __align__(32) _Float16 wlds[EDGE_WLDS_HALFS];  // 28KB fc weights
  __shared__ __align__(32) _Float16 sh1[4][16 * 64];
  __shared__ __align__(32) _Float16 sh2[4][16 * 64];
  __shared__ __align__(16) float    sw [4][16 * 128];

  int lane = threadIdx.x & 31;
  int wid  = threadIdx.x >> 5;
  int tile = blockIdx.x * 4 + wid;

  stage_weights(wlds, pk, OFF_FCW0, EDGE_WLDS_HALFS, wid);
  if (tile * 16 >= N_EDGES) return;

  int r = lane & 15;
  int half = lane >> 4;
  int h8 = half * 8;

  const float* h0t = ws + H0_F;
  const float* h1t = ws + H1_F;
  float* s0t = ws + S0_F;
  float* s1t = ws + S1_F;

  // ---- tensor-product lane mapping, resolved early so we can prefetch ----
  int eloc = lane >> 1;           // edge within tile (0..15)
  int uh   = (lane & 1) * 16;     // channel half (0 or 16)
  int e    = tile * 16 + eloc;
  int src  = esrc[e];
  int dst  = edst[e];
  const float* h0p = h0t + (size_t)src * 32;
  const float* h1p = h1t + (size_t)src * 96;
  // hide random-gather latency behind the WMMA MLP
  __builtin_prefetch(h0p, 0, 3);
  __builtin_prefetch(h1p, 0, 3);
  __builtin_prefetch(h1p + 32, 0, 3);
  __builtin_prefetch(h1p + 64, 0, 3);

  // ---- layer 1: emb(16x8, K padded to 32) @ fc_w0 -> 16x64, silu ----
  int e_r = tile * 16 + r;
  const float* er = emb + (size_t)e_r * 8;
  v16h a1;
#pragma unroll
  for (int i = 0; i < 16; ++i) {
    int kk = kkmap(i, h8);
    a1[i] = (kk < 8) ? (_Float16)er[kk] : (_Float16)0.f;
  }
  _Float16* H1 = sh1[wid];
#pragma unroll
  for (int nt = 0; nt < 4; ++nt) {
    v8f acc = wmma32(a1, ldB(wlds, OFF_FCW0 - OFF_FCW0, nt, lane), vzero8f());
#pragma unroll
    for (int i = 0; i < 8; ++i)
      H1[(i + h8) * 64 + nt * 16 + r] = (_Float16)silu(acc[i]);
  }
  wave_lds_fence();

  // ---- layer 2: 16x64 @ fc_w1 -> 16x64, silu ----
  _Float16* H2 = sh2[wid];
#pragma unroll
  for (int nt = 0; nt < 4; ++nt) {
    v8f acc = vzero8f();
#pragma unroll
    for (int kt = 0; kt < 2; ++kt) {
      v8h lo = *(const v8h*)(H1 + r * 64 + kt * 32 + h8);
      v8h hi = *(const v8h*)(H1 + r * 64 + kt * 32 + 16 + h8);
      v16h a;
#pragma unroll
      for (int i = 0; i < 8; ++i) { a[i] = lo[i]; a[8 + i] = hi[i]; }
      acc = wmma32(a, ldB(wlds, OFF_FCW1 - OFF_FCW0, kt * 4 + nt, lane), acc);
    }
#pragma unroll
    for (int i = 0; i < 8; ++i)
      H2[(i + h8) * 64 + nt * 16 + r] = (_Float16)silu(acc[i]);
  }
  wave_lds_fence();

  // ---- layer 3: 16x64 @ fc_w2 -> 16x128 (linear) ----
  float* W = sw[wid];
#pragma unroll
  for (int nt = 0; nt < 8; ++nt) {
    v8f acc = vzero8f();
#pragma unroll
    for (int kt = 0; kt < 2; ++kt) {
      v8h lo = *(const v8h*)(H2 + r * 64 + kt * 32 + h8);
      v8h hi = *(const v8h*)(H2 + r * 64 + kt * 32 + 16 + h8);
      v16h a;
#pragma unroll
      for (int i = 0; i < 8; ++i) { a[i] = lo[i]; a[8 + i] = hi[i]; }
      acc = wmma32(a, ldB(wlds, OFF_FCW2 - OFF_FCW0, kt * 8 + nt, lane), acc);
    }
#pragma unroll
    for (int i = 0; i < 8; ++i)
      W[(i + h8) * 128 + nt * 16 + r] = acc[i];
  }
  wave_lds_fence();

  // ---- tensor product + segment-sum scatter (2 lanes per edge) ----
  const float* shp = esh + (size_t)e * 4;
  float y0  = shp[0];
  float y1x = shp[1], y1y = shp[2], y1z = shp[3];
  float* s0p = s0t + (size_t)dst * 64;
  float* s1p = s1t + (size_t)dst * 192;
  const float* wrow = W + eloc * 128;
  const float INV_NB = 1.0f / 16.0f;
  const float IS3    = 0.5773502691896258f;
#pragma unroll
  for (int uu = 0; uu < 16; ++uu) {
    int u = uh + uu;
    float w0 = wrow[u], w1 = wrow[32 + u], w2 = wrow[64 + u], w3 = wrow[96 + u];
    float e0  = h0p[u];
    float e1x = h1p[u * 3], e1y = h1p[u * 3 + 1], e1z = h1p[u * 3 + 2];
    float dotv = e1x * y1x + e1y * y1y + e1z * y1z;
    atomAddF(s0p + u,      INV_NB * w0 * e0 * y0);
    atomAddF(s0p + 32 + u, INV_NB * IS3 * w3 * dotv);
    float t1 = INV_NB * w1 * e0;
    atomAddF(s1p + u * 3 + 0, t1 * y1x);
    atomAddF(s1p + u * 3 + 1, t1 * y1y);
    atomAddF(s1p + u * 3 + 2, t1 * y1z);
    float t2 = INV_NB * w2 * y0;
    atomAddF(s1p + (32 + u) * 3 + 0, t2 * e1x);
    atomAddF(s1p + (32 + u) * 3 + 1, t2 * e1y);
    atomAddF(s1p + (32 + u) * 3 + 2, t2 * e1z);
  }
}

// ======================== final: lin2 + gate ==============================
__global__ __launch_bounds__(128) void final_kernel(
    const _Float16* __restrict__ pk, const float* __restrict__ ws,
    float* __restrict__ out) {
  __shared__ __align__(32) _Float16 wlds[FIN_WLDS_HALFS];   // 12KB l2 weights
  int lane = threadIdx.x & 31;
  int wid  = threadIdx.x >> 5;
  int tile = blockIdx.x * 4 + wid;

  stage_weights(wlds, pk, OFF_L2W0, FIN_WLDS_HALFS, wid);
  if (tile * 16 >= N_NODES) return;

  int r = lane & 15;
  int half = lane >> 4;
  int h8 = half * 8;
  int base = tile * 16;

  const float* s0  = ws + S0_F;
  const float* s1  = ws + S1_F;
  const float* scs = ws + SCS_F;
  const float* scg = ws + SCG_F;
  const float* scv = ws + SCV_F;

  // o0 = s0 @ lin2_w0 (inv_nb & l2 folded into weights)
  v8f o0[4];
#pragma unroll
  for (int nt = 0; nt < 4; ++nt) o0[nt] = vzero8f();
  const float* s0r = s0 + (size_t)(base + r) * 64;
#pragma unroll
  for (int kt = 0; kt < 2; ++kt) {
    v16h a;
#pragma unroll
    for (int i = 0; i < 16; ++i)
      a[i] = (_Float16)s0r[kt * 32 + kkmap(i, h8)];
#pragma unroll
    for (int nt = 0; nt < 4; ++nt)
      o0[nt] = wmma32(a, ldB(wlds, 0, kt * 4 + nt, lane), o0[nt]);
  }

  // o1[m] = s1[:,:,m] @ lin2_w1
  v8f o1[3][2];
  const float* s1r = s1 + (size_t)(base + r) * 192;
  for (int m = 0; m < 3; ++m) {
    o1[m][0] = vzero8f(); o1[m][1] = vzero8f();
#pragma unroll
    for (int kt = 0; kt < 2; ++kt) {
      v16h a;
#pragma unroll
      for (int i = 0; i < 16; ++i)
        a[i] = (_Float16)s1r[(kt * 32 + kkmap(i, h8)) * 3 + m];
#pragma unroll
      for (int nt = 0; nt < 2; ++nt)
        o1[m][nt] = wmma32(a, ldB(wlds, OFF_L2W1 - OFF_L2W0, kt * 2 + nt, lane),
                           o1[m][nt]);
    }
  }

  // combine with self-connection, gate, write (N,128) output
#pragma unroll
  for (int nt = 0; nt < 2; ++nt) {
    int col = nt * 16 + r;
#pragma unroll
    for (int i = 0; i < 8; ++i) {
      int row = base + i + h8;
      float sscal = o0[nt][i] + scs[(size_t)row * 32 + col];
      out[(size_t)row * 128 + col] = silu(sscal);
      float g = silu(o0[2 + nt][i] + scg[(size_t)row * 32 + col]);
#pragma unroll
      for (int m = 0; m < 3; ++m) {
        float v = o1[m][nt][i] + scv[(size_t)row * 96 + col * 3 + m];
        out[(size_t)row * 128 + 32 + col * 3 + m] = g * v;
      }
    }
  }
}

// ======================== launch ==========================================
extern "C" void kernel_launch(void* const* d_in, const int* in_sizes, int n_in,
                              void* d_out, int out_size, void* d_ws,
                              size_t ws_size, hipStream_t stream) {
  (void)in_sizes; (void)n_in; (void)out_size; (void)ws_size;
  const float* nf   = (const float*)d_in[0];
  const float* z    = (const float*)d_in[1];
  const float* esh  = (const float*)d_in[2];
  const float* emb  = (const float*)d_in[3];
  const float* l1w0 = (const float*)d_in[4];
  const float* l1w1 = (const float*)d_in[5];
  const float* fcw0 = (const float*)d_in[6];
  const float* fcw1 = (const float*)d_in[7];
  const float* fcw2 = (const float*)d_in[8];
  const float* scws = (const float*)d_in[9];
  const float* scwg = (const float*)d_in[10];
  const float* scwv = (const float*)d_in[11];
  const float* l2w0 = (const float*)d_in[12];
  const float* l2w1 = (const float*)d_in[13];
  const int*   esrc = (const int*)d_in[14];
  const int*   edst = (const int*)d_in[15];

  _Float16* pk = (_Float16*)d_ws;
  float* wsf = (float*)((char*)d_ws + (size_t)PK_HALFS * 2);
  float* out = (float*)d_out;

  // zero the segment-sum accumulators (s0: N*64, s1: N*192 floats, contiguous)
  hipMemsetAsync((char*)d_ws + (size_t)PK_HALFS * 2 + (size_t)S0_F * 4, 0,
                 (size_t)N_NODES * 256 * 4, stream);

  pack_weights_kernel<<<64, 256, 0, stream>>>(l1w0, l1w1, fcw0, fcw1, fcw2,
                                              scws, scwg, scwv, l2w0, l2w1, pk);

  int node_tiles = (N_NODES + 15) / 16;               // 3125
  int node_blocks = (node_tiles + 3) / 4;             // 782
  node_prep_kernel<<<node_blocks, 128, 0, stream>>>(nf, z, pk, wsf);

  int edge_blocks = (N_EDGES / 16) / 4;               // 12500
  edge_kernel<<<edge_blocks, 128, 0, stream>>>(esh, emb, esrc, edst, pk, wsf);

  final_kernel<<<node_blocks, 128, 0, stream>>>(pk, wsf, out);
}
